// ConvBlock_17927193493940
// MI455X (gfx1250) — 8-bit matrix operand order repaired and checked by reading, compiled, not run on hardware
//
#include <hip/hip_runtime.h>


typedef __attribute__((ext_vector_type(8))) int v8i;

#define NB      32
#define CH      16
#define HH      224
#define WW      224
#define HWC     (HH * WW)
#define NPIX    (NB * HWC)
#define NTOT    (NB * CH * HWC)
#define NTOT4   (NTOT / 4)
#define KTOT    144
#define KPAD    192
#define NBLK    (NB * HH)
#define MELEM   1605632.0f
#define BN_EPS  1e-5f
#define ACT_TH  6.0f

#define WS_SUM_OFF    64
#define WS_SUMSQ_OFF  128
#define WS_WQ_OFF     256
#define WS_PART_OFF   4096
#define WS_XQ_OFF     (4096 + NBLK * 32 * 4)

__global__ void k_init(unsigned* hdr) {
    if (threadIdx.x < 64) hdr[threadIdx.x] = 0u;
}

__global__ void k_absmax_x(const float* __restrict__ x, unsigned* __restrict__ hdr) {
    __shared__ float red[256];
    const int tid = threadIdx.x;
    float m = 0.0f;
    for (int i = blockIdx.x * blockDim.x + tid; i < NTOT4; i += gridDim.x * blockDim.x) {
        float4 v = reinterpret_cast<const float4*>(x)[i];
        m = fmaxf(m, fabsf(v.x));
        m = fmaxf(m, fabsf(v.y));
        m = fmaxf(m, fabsf(v.z));
        m = fmaxf(m, fabsf(v.w));
    }
    red[tid] = m;
    __syncthreads();
    for (int s = 128; s > 0; s >>= 1) {
        if (tid < s) red[tid] = fmaxf(red[tid], red[tid + s]);
        __syncthreads();
    }
    if (tid == 0) atomicMax(&hdr[0], __float_as_uint(red[0]));
}

__global__ void k_prep_w(const float* __restrict__ w, unsigned* __restrict__ hdr,
                         signed char* __restrict__ wqp) {
    __shared__ float red[256];
    const int tid = threadIdx.x;
    float m = 0.0f;
    for (int i = tid; i < CH * KTOT; i += 256) m = fmaxf(m, fabsf(w[i]));
    red[tid] = m;
    __syncthreads();
    for (int s = 128; s > 0; s >>= 1) {
        if (tid < s) red[tid] = fmaxf(red[tid], red[tid + s]);
        __syncthreads();
    }
    const float aw = red[0];
    if (tid == 0) hdr[1] = __float_as_uint(aw);
    const float invstep = (aw > 0.0f) ? (127.0f / aw) : 0.0f;
    for (int i = tid; i < CH * KPAD; i += 256) {
        const int cout = i / KPAD;
        const int k    = i - cout * KPAD;
        signed char q = 0;
        if (k < KTOT) {
            const int tap = k >> 4;
            const int cin = k & 15;
            int qi = __float2int_rn(w[(cout * CH + cin) * 9 + tap] * invstep);
            qi = max(-127, min(127, qi));
            q = (signed char)qi;
        }
        wqp[i] = q;
    }
}

__global__ void k_quant_x(const float* __restrict__ x, unsigned* __restrict__ xqt,
                          const unsigned* __restrict__ hdr) {
    const int pix = blockIdx.x * blockDim.x + threadIdx.x;
    if (pix >= NPIX) return;
    const int n  = pix / HWC;
    const int hw = pix - n * HWC;
    const float ax = __uint_as_float(hdr[0]);
    const float invstep = (ax > 0.0f) ? (127.0f / ax) : 0.0f;
    unsigned packed[4] = {0u, 0u, 0u, 0u};
#pragma unroll
    for (int c = 0; c < CH; ++c) {
        const float v = x[(n * CH + c) * HWC + hw];
        int q = max(-127, min(127, __float2int_rn(v * invstep)));
        packed[c >> 2] |= (unsigned)(q & 0xff) << (8 * (c & 3));
    }
    uint4 o; o.x = packed[0]; o.y = packed[1]; o.z = packed[2]; o.w = packed[3];
    reinterpret_cast<uint4*>(xqt)[pix] = o;
}

#define TILE_COLS 226
#define TILE_CSTR 228
#define SM_WQ_OFF  (3 * TILE_CSTR * 16)
#define SM_ACC_OFF (SM_WQ_OFF + CH * KPAD)

__global__ void __launch_bounds__(448)
k_conv_wmma(const signed char* __restrict__ xqt, const signed char* __restrict__ wqp,
            const unsigned* __restrict__ hdr, float* __restrict__ out,
            float* __restrict__ partials) {
    __shared__ unsigned char smem[SM_ACC_OFF + 32 * 4];

    const int tid  = threadIdx.x;
    const int lane = tid & 31;
    const int wv   = tid >> 5;
    const int n    = blockIdx.x / HH;
    const int h    = blockIdx.x - n * HH;
    float* accs = reinterpret_cast<float*>(&smem[SM_ACC_OFF]);

    if (tid < 32) accs[tid] = 0.0f;

    for (int i = tid; i < 3 * TILE_COLS; i += 448) {
        const int r   = i / TILE_COLS;
        const int col = i - r * TILE_COLS;
        const int gh  = h - 1 + r;
        const int gw  = col - 1;
        uint4 v = make_uint4(0u, 0u, 0u, 0u);
        if ((unsigned)gh < (unsigned)HH && (unsigned)gw < (unsigned)WW)
            v = reinterpret_cast<const uint4*>(xqt)[(n * HH + gh) * WW + gw];
        *reinterpret_cast<uint4*>(&smem[(r * TILE_CSTR + col) * 16]) = v;
    }
    for (int i = tid; i < (CH * KPAD) / 16; i += 448)
        *reinterpret_cast<uint4*>(&smem[SM_WQ_OFF + i * 16]) =
            reinterpret_cast<const uint4*>(wqp)[i];
    __syncthreads();

    const int p   = lane & 15;
    const int hi  = lane >> 4;
    const int px  = wv * 16 + p;

    v8i acc = {0, 0, 0, 0, 0, 0, 0, 0};

#pragma unroll
    for (int kc = 0; kc < 3; ++kc) {
        v8i a;
#pragma unroll
        for (int j = 0; j < 8; ++j) {
            const int kbase = kc * 64 + ((j >> 2) * 32 + (j & 3) * 4) + hi * 16;
            a[j] = *reinterpret_cast<const int*>(&smem[SM_WQ_OFF + p * KPAD + kbase]);
        }

        v8i b;
#pragma unroll
        for (int g = 0; g < 2; ++g) {
            const int t = 4 * kc + hi + 2 * g;
            uint4 bv = make_uint4(0u, 0u, 0u, 0u);
            if (t < 9) {
                const int r = t / 3;
                const int s = t - 3 * r;
                const int col = px + s;
                bv = *reinterpret_cast<const uint4*>(&smem[(r * TILE_CSTR + col) * 16]);
            }
            b[g * 4 + 0] = (int)bv.x;
            b[g * 4 + 1] = (int)bv.y;
            b[g * 4 + 2] = (int)bv.z;
            b[g * 4 + 3] = (int)bv.w;
        }

        acc = __builtin_amdgcn_wmma_i32_16x16x64_iu8(true, a, true, b, acc, false, false);
    }

    const float ax = __uint_as_float(hdr[0]);
    const float aw = __uint_as_float(hdr[1]);
    const float scale = (ax * (1.0f / 127.0f)) * (aw * (1.0f / 127.0f));
#pragma unroll
    for (int j = 0; j < 8; ++j) {
        const int cout = j + hi * 8;
        const float v = (float)acc[j] * scale;
        out[((n * CH + cout) * HH + h) * WW + px] = v;
        float vs = v, vq = v * v;
#pragma unroll
        for (int m = 1; m <= 8; m <<= 1) {
            vs += __shfl_xor(vs, m, 32);
            vq += __shfl_xor(vq, m, 32);
        }
        if (p == 0) {
            atomicAdd(&accs[cout], vs);
            atomicAdd(&accs[16 + cout], vq);
        }
    }
    __syncthreads();
    if (tid < 32) partials[(size_t)blockIdx.x * 32 + tid] = accs[tid];
}

__global__ void k_reduce(const float* __restrict__ partials,
                         float* __restrict__ sums, float* __restrict__ sumsq) {
    __shared__ float red[256];
    const int tid = threadIdx.x;
    const int v = blockIdx.x;
    float s = 0.0f;
    for (int b = tid; b < NBLK; b += 256) s += partials[(size_t)b * 32 + v];
    red[tid] = s;
    __syncthreads();
    for (int off = 128; off > 0; off >>= 1) {
        if (tid < off) red[tid] += red[tid + off];
        __syncthreads();
    }
    if (tid == 0) {
        if (v < 16) sums[v] = red[0];
        else        sumsq[v - 16] = red[0];
    }
}

__global__ void k_bn_apply(float* __restrict__ y, const float* __restrict__ sums,
                           const float* __restrict__ sumsq,
                           const float* __restrict__ gamma,
                           const float* __restrict__ beta) {
    const int i4 = blockIdx.x * blockDim.x + threadIdx.x;
    if (i4 >= NTOT4) return;
    const int c = (i4 / (HWC / 4)) & 15;
    const float mean = sums[c] * (1.0f / MELEM);
    const float var  = fmaxf(sumsq[c] * (1.0f / MELEM) - mean * mean, 0.0f);
    const float inv  = gamma[c] * rsqrtf(var + BN_EPS);
    const float bias = beta[c] - mean * inv;
    float4 v = reinterpret_cast<float4*>(y)[i4];
    v.x = fminf(fmaxf(v.x * inv + bias, 0.0f), ACT_TH);
    v.y = fminf(fmaxf(v.y * inv + bias, 0.0f), ACT_TH);
    v.z = fminf(fmaxf(v.z * inv + bias, 0.0f), ACT_TH);
    v.w = fminf(fmaxf(v.w * inv + bias, 0.0f), ACT_TH);
    reinterpret_cast<float4*>(y)[i4] = v;
}

extern "C" void kernel_launch(void* const* d_in, const int* in_sizes, int n_in,
                              void* d_out, int out_size, void* d_ws, size_t ws_size,
                              hipStream_t stream) {
    (void)in_sizes; (void)n_in; (void)out_size; (void)ws_size;
    const float* x      = (const float*)d_in[0];
    const float* weight = (const float*)d_in[1];
    const float* gamma  = (const float*)d_in[2];
    const float* beta   = (const float*)d_in[3];
    float* out = (float*)d_out;

    unsigned*    hdr      = (unsigned*)d_ws;
    float*       sums     = (float*)((char*)d_ws + WS_SUM_OFF);
    float*       sumsq    = (float*)((char*)d_ws + WS_SUMSQ_OFF);
    signed char* wqp      = (signed char*)d_ws + WS_WQ_OFF;
    float*       partials = (float*)((char*)d_ws + WS_PART_OFF);
    signed char* xqt      = (signed char*)d_ws + WS_XQ_OFF;

    k_init<<<1, 64, 0, stream>>>(hdr);
    k_absmax_x<<<2048, 256, 0, stream>>>(x, hdr);
    k_prep_w<<<1, 256, 0, stream>>>(weight, hdr, wqp);
    k_quant_x<<<NPIX / 256, 256, 0, stream>>>(x, (unsigned*)xqt, hdr);
    k_conv_wmma<<<NBLK, 448, 0, stream>>>(xqt, wqp, hdr, out, partials);
    k_reduce<<<32, 256, 0, stream>>>(partials, sums, sumsq);
    k_bn_apply<<<NTOT4 / 256, 256, 0, stream>>>(out, sums, sumsq, gamma, beta);
}
